// Cross_attention_69389491634474
// MI455X (gfx1250) — compile-verified, matrix operand order checked by reading, not run on hardware
//
#include <hip/hip_runtime.h>
#include <hip/hip_bf16.h>

// ---------------------------------------------------------------------------
// Cross-attention pipeline for MI455X (gfx1250), wave32 + WMMA bf16.
// All GEMMs are NT (both operands K-contiguous) -> per-lane v16bf fragment is
// one contiguous 32B global load feeding v_wmma_f32_16x16x32_bf16.
// Wave tile 32x64 (2x4 WMMA), block tile 64x256 (8 waves).
// K loop: outer 256-element chunks (prefetch once, branch-free), inner
// unroll-8 with immediate-offset loads.
// ---------------------------------------------------------------------------

typedef __attribute__((ext_vector_type(16))) __bf16 v16bf;
typedef __attribute__((ext_vector_type(8)))  float  v8f;

constexpr int Bq = 16;    // batch
constexpr int Sd = 2048;  // doc length
constexpr int LQ = 512;   // query length
constexpr int Hd = 1024;  // hidden

__device__ __forceinline__ unsigned short f2bf(float x) {
  unsigned int u = __float_as_uint(x);
  u += 0x7FFFu + ((u >> 16) & 1u);   // round-to-nearest-even
  return (unsigned short)(u >> 16);
}

__device__ __forceinline__ v16bf ldbf16x16(const unsigned short* p) {
  return *reinterpret_cast<const v16bf*>(p);
}

// ---------------------------------------------------------------------------
// Generic batched NT GEMM: C[m,n] = sum_k A[m,k] * B[n,k] (+ bias[n])
// A,B bf16 (K contiguous), C fp32 or bf16. Optional split-A: rows are the
// concatenation [A (splitK cols) | A2 (K-splitK cols)].
// Block: 256 threads = 8 waves (2Mx4N); wave tile 32x64; block tile 64x256.
// Requires: K % 256 == 0, splitK % 256 == 0 (true for all uses here).
// ---------------------------------------------------------------------------
template<bool OUT_BF16, bool HAS_BIAS, bool SPLIT_A>
__global__ __launch_bounds__(256) void gemm_nt(
    const unsigned short* __restrict__ A,  long long aBatch,  int lda,
    const unsigned short* __restrict__ A2, long long aBatch2, int lda2, int splitK,
    const unsigned short* __restrict__ Bm, long long bBatch,  int ldb,
    float* __restrict__ Cf, unsigned short* __restrict__ Ch,
    long long cBatch, int ldc,
    const float* __restrict__ bias, int K)
{
  const int b    = blockIdx.z;
  const int lane = threadIdx.x & 31;
  const int wave = threadIdx.x >> 5;
  const int l16  = lane & 15;
  const int hsel = lane >> 4;           // 0: K 0..15, 1: K 16..31 of a step

  const int mBase = blockIdx.x * 64  + (wave & 1) * 32;
  const int nBase = blockIdx.y * 256 + (wave >> 1) * 64;

  const unsigned short* Ab  = A  + (long long)b * aBatch;
  const unsigned short* A2b = SPLIT_A ? (A2 + (long long)b * aBatch2) : nullptr;
  const unsigned short* Bb  = Bm + (long long)b * bBatch;

  v8f acc[2][4] = {};

  const int ar0 = mBase + l16;
  const int ar1 = ar0 + 16;
  const int hk  = hsel * 16;            // lane's K sub-offset within a step

  for (int kc = 0; kc < K; kc += 256) {
    // Chunk base pointers (split-A select is chunk-uniform: splitK%256==0).
    const unsigned short *pa0, *pa1;
    if (!SPLIT_A || kc < splitK) {
      pa0 = Ab + (long long)ar0 * lda + kc + hk;
      pa1 = Ab + (long long)ar1 * lda + kc + hk;
    } else {
      const int k2 = kc - splitK;
      pa0 = A2b + (long long)ar0 * lda2 + k2 + hk;
      pa1 = A2b + (long long)ar1 * lda2 + k2 + hk;
    }
    const unsigned short* pb[4];
#pragma unroll
    for (int j = 0; j < 4; ++j)
      pb[j] = Bb + (long long)(nBase + j * 16 + l16) * ldb + kc + hk;

    // One speculative prefetch per operand row per 256-element chunk
    // (branch-free; lowers to global_prefetch_b8). Next chunk = +512B.
    __builtin_prefetch(pa0 + 256, 0, 1);
    __builtin_prefetch(pa1 + 256, 0, 1);
#pragma unroll
    for (int j = 0; j < 4; ++j) __builtin_prefetch(pb[j] + 256, 0, 1);

    // 8 x (6 loads + 8 WMMAs), all loads at immediate offsets u*64 bytes.
#pragma unroll
    for (int u = 0; u < 8; ++u) {
      const int ko = u * 32;
      const v16bf a0 = ldbf16x16(pa0 + ko);
      const v16bf a1 = ldbf16x16(pa1 + ko);
      v16bf bf[4];
#pragma unroll
      for (int j = 0; j < 4; ++j) bf[j] = ldbf16x16(pb[j] + ko);
#pragma unroll
      for (int j = 0; j < 4; ++j) {
        acc[0][j] = __builtin_amdgcn_wmma_f32_16x16x32_bf16(false, a0, false, bf[j], (short)0, acc[0][j], false, false);
        acc[1][j] = __builtin_amdgcn_wmma_f32_16x16x32_bf16(false, a1, false, bf[j], (short)0, acc[1][j], false, false);
      }
    }
  }

  float* Cfb          = OUT_BF16 ? nullptr : (Cf + (long long)b * cBatch);
  unsigned short* Chb = OUT_BF16 ? (Ch + (long long)b * cBatch) : nullptr;

#pragma unroll
  for (int mt = 0; mt < 2; ++mt) {
#pragma unroll
    for (int j = 0; j < 4; ++j) {
      const v8f a = acc[mt][j];
      const int col = nBase + j * 16 + l16;
      float bv = 0.0f;
      if constexpr (HAS_BIAS) bv = bias[col];
#pragma unroll
      for (int r = 0; r < 8; ++r) {
        // C/D layout: lanes 0-15 rows m=r, lanes 16-31 rows m=8+r
        const int row = mBase + mt * 16 + hsel * 8 + r;
        float v = a[r];
        if constexpr (HAS_BIAS) v += bv;
        const long long idx = (long long)row * ldc + col;
        if constexpr (OUT_BF16) Chb[idx] = f2bf(v);
        else                    Cfb[idx] = v;
      }
    }
  }
}

// ---------------------------------------------------------------------------
// LDS-tiled bf16 transpose: in (R x C) -> out (C x R), batched.
// ---------------------------------------------------------------------------
__global__ __launch_bounds__(256) void transpose16(
    const unsigned short* __restrict__ in, unsigned short* __restrict__ out,
    int R, int C, long long inBatch, long long outBatch)
{
  __shared__ unsigned short tile[32][33];
  const int b  = blockIdx.z;
  const int r0 = blockIdx.y * 32;
  const int c0 = blockIdx.x * 32;
  const int tx = threadIdx.x & 31;
  const int ty = threadIdx.x >> 5;
  const unsigned short* ib = in + (long long)b * inBatch;
  unsigned short* ob       = out + (long long)b * outBatch;
#pragma unroll
  for (int i = ty; i < 32; i += 8)
    tile[i][tx] = ib[(long long)(r0 + i) * C + (c0 + tx)];
  __syncthreads();
#pragma unroll
  for (int i = ty; i < 32; i += 8)
    ob[(long long)(c0 + i) * R + (r0 + tx)] = tile[tx][i];
}

// ---------------------------------------------------------------------------
// Elementwise fp32 -> bf16 casts.
// ---------------------------------------------------------------------------
__global__ void cast_bf16(const float* __restrict__ in,
                          unsigned short* __restrict__ out, long long n) {
  const long long i = (long long)blockIdx.x * blockDim.x + threadIdx.x;
  if (i < n) out[i] = f2bf(in[i]);
}

// Q (B,LQ,H) fp32 -> columns [0,H) of QOh (B,LQ,2H) bf16
__global__ void pack_q(const float* __restrict__ Q,
                       unsigned short* __restrict__ QOh, long long n) {
  const long long i = (long long)blockIdx.x * blockDim.x + threadIdx.x;
  if (i < n) {
    const int h = (int)(i % Hd);
    const long long r = i / Hd;           // b*LQ + q
    QOh[r * (2 * Hd) + h] = f2bf(Q[i]);
  }
}

// ---------------------------------------------------------------------------
// Row softmax over LQ with query mask -> A_d bf16 (B,S,LQ). One block/row.
// ---------------------------------------------------------------------------
__global__ __launch_bounds__(256) void softmax_ad(
    const float* __restrict__ scores, const unsigned char* __restrict__ qmask,
    unsigned short* __restrict__ Ad)
{
  const int s = blockIdx.x, b = blockIdx.y;
  const long long base = ((long long)b * Sd + s) * LQ;
  const int t = threadIdx.x;
  const float NEG = -__builtin_inff();
  const float v0 = qmask[base + t]       ? NEG : scores[base + t];
  const float v1 = qmask[base + t + 256] ? NEG : scores[base + t + 256];

  __shared__ float red[256];
  red[t] = fmaxf(v0, v1);
  __syncthreads();
  for (int o = 128; o > 0; o >>= 1) { if (t < o) red[t] = fmaxf(red[t], red[t + o]); __syncthreads(); }
  const float m = red[0];
  __syncthreads();

  const float e0 = __expf(v0 - m), e1 = __expf(v1 - m);
  red[t] = e0 + e1;
  __syncthreads();
  for (int o = 128; o > 0; o >>= 1) { if (t < o) red[t] += red[t + o]; __syncthreads(); }
  const float inv = 1.0f / red[0];

  Ad[base + t]       = f2bf(e0 * inv);
  Ad[base + t + 256] = f2bf(e1 * inv);
}

// ---------------------------------------------------------------------------
// Column softmax over S with doc mask -> A_q bf16 (B,LQ,S). One block/(b,q).
// Strided score reads stay L2-resident (4.2MB/batch slab, 192MB L2).
// ---------------------------------------------------------------------------
__global__ __launch_bounds__(256) void softmax_aq(
    const float* __restrict__ scores, const unsigned char* __restrict__ dmask,
    unsigned short* __restrict__ Aq)
{
  const int q = blockIdx.x, b = blockIdx.y;
  const float* col         = scores + (long long)b * Sd * LQ + q;
  const unsigned char* dm  = dmask + ((long long)b * LQ + q) * Sd;
  unsigned short* op       = Aq    + ((long long)b * LQ + q) * Sd;
  const int t = threadIdx.x;
  const float NEG = -__builtin_inff();

  float v[8];
  float m = NEG;
#pragma unroll
  for (int i = 0; i < 8; ++i) {
    const int s = t + i * 256;
    v[i] = dm[s] ? NEG : col[(long long)s * LQ];
    m = fmaxf(m, v[i]);
  }
  __shared__ float red[256];
  red[t] = m;
  __syncthreads();
  for (int o = 128; o > 0; o >>= 1) { if (t < o) red[t] = fmaxf(red[t], red[t + o]); __syncthreads(); }
  m = red[0];
  __syncthreads();

  float e[8], sum = 0.0f;
#pragma unroll
  for (int i = 0; i < 8; ++i) { e[i] = __expf(v[i] - m); sum += e[i]; }
  red[t] = sum;
  __syncthreads();
  for (int o = 128; o > 0; o >>= 1) { if (t < o) red[t] += red[t + o]; __syncthreads(); }
  const float inv = 1.0f / red[0];
#pragma unroll
  for (int i = 0; i < 8; ++i) op[t + i * 256] = f2bf(e[i] * inv);
}

// last valid doc position gather -> (1,B,H)
__global__ void extract_last(const float* __restrict__ doc_out,
                             const int* __restrict__ doc_lens,
                             float* __restrict__ out_last) {
  const int i = blockIdx.x * blockDim.x + threadIdx.x;   // B*H threads
  const int b = i / Hd, h = i % Hd;
  const int pos = doc_lens[b] - 1;
  out_last[i] = doc_out[((long long)b * Sd + pos) * Hd + h];
}

// ---------------------------------------------------------------------------
extern "C" void kernel_launch(void* const* d_in, const int* in_sizes, int n_in,
                              void* d_out, int out_size, void* d_ws, size_t ws_size,
                              hipStream_t stream) {
  const float* D  = (const float*)d_in[0];
  const float* Q  = (const float*)d_in[1];
  const float* W1 = (const float*)d_in[2];
  const float* b1 = (const float*)d_in[3];
  const float* W2 = (const float*)d_in[4];
  const float* b2 = (const float*)d_in[5];
  const unsigned char* qmask = (const unsigned char*)d_in[6];  // (B,S,LQ) bool
  const unsigned char* dmask = (const unsigned char*)d_in[7];  // (B,LQ,S) bool
  const int* doc_lens = (const int*)d_in[8];

  float* out_doc  = (float*)d_out;                       // (B,S,H)
  float* out_last = out_doc + (size_t)Bq * Sd * Hd;      // (1,B,H)
  float* out_q    = out_last + (size_t)Bq * Hd;          // (B,LQ,H)

  char* ws = (char*)d_ws;
  size_t off = 0;
  auto alloc = [&](size_t bytes) -> void* {
    void* p = ws + off;
    off = (off + bytes + 255) & ~(size_t)255;
    return p;
  };
  unsigned short* Dh  = (unsigned short*)alloc((size_t)Bq * Sd * Hd * 2);      // bf16 D
  unsigned short* Qh  = (unsigned short*)alloc((size_t)Bq * LQ * Hd * 2);      // bf16 Q
  unsigned short* DhT = (unsigned short*)alloc((size_t)Bq * Sd * Hd * 2);      // bf16 D^T (H,S)
  unsigned short* W1h = (unsigned short*)alloc((size_t)3 * Hd * Hd * 2);
  unsigned short* W2h = (unsigned short*)alloc((size_t)2 * Hd * Hd * 2);
  float*          Sc  = (float*)alloc((size_t)Bq * Sd * LQ * 4);               // fp32 scores
  unsigned short* Ad  = (unsigned short*)alloc((size_t)Bq * Sd * LQ * 2);      // bf16 A_d
  unsigned short* Aq  = (unsigned short*)alloc((size_t)Bq * LQ * Sd * 2);      // bf16 A_q
  unsigned short* QOh = (unsigned short*)alloc((size_t)Bq * LQ * 2 * Hd * 2);  // bf16 [Q|C_q]
  unsigned short* QOT = (unsigned short*)alloc((size_t)Bq * 2 * Hd * LQ * 2);  // bf16 QO^T
  unsigned short* Cd  = (unsigned short*)alloc((size_t)Bq * Sd * 2 * Hd * 2);  // bf16 C_D

  // --- casts ---
  { long long n = (long long)Bq * Sd * Hd; cast_bf16<<<(int)((n + 255) / 256), 256, 0, stream>>>(D, Dh, n); }
  { long long n = (long long)Bq * LQ * Hd; cast_bf16<<<(int)((n + 255) / 256), 256, 0, stream>>>(Q, Qh, n); }
  { long long n = (long long)3 * Hd * Hd;  cast_bf16<<<(int)((n + 255) / 256), 256, 0, stream>>>(W1, W1h, n); }
  { long long n = (long long)2 * Hd * Hd;  cast_bf16<<<(int)((n + 255) / 256), 256, 0, stream>>>(W2, W2h, n); }
  { long long n = (long long)Bq * LQ * Hd; pack_q<<<(int)((n + 255) / 256), 256, 0, stream>>>(Q, QOh, n); }

  // D^T (per-batch S x H -> H x S)
  transpose16<<<dim3(Hd / 32, Sd / 32, Bq), 256, 0, stream>>>(
      Dh, DhT, Sd, Hd, (long long)Sd * Hd, (long long)Sd * Hd);

  // scores(S,LQ) = Dh(S,H) x Qh(LQ,H)^T, fp32
  gemm_nt<false, false, false><<<dim3(Sd / 64, LQ / 256, Bq), 256, 0, stream>>>(
      Dh, (long long)Sd * Hd, Hd,
      nullptr, 0, 0, 0,
      Qh, (long long)LQ * Hd, Hd,
      Sc, nullptr, (long long)Sd * LQ, LQ,
      nullptr, Hd);

  softmax_ad<<<dim3(Sd, Bq), 256, 0, stream>>>(Sc, qmask, Ad);
  softmax_aq<<<dim3(LQ, Bq), 256, 0, stream>>>(Sc, dmask, Aq);

  // C_q(LQ,H) = Aq(LQ,S) x DhT(H,S)^T -> bf16 into QOh columns [H,2H)
  gemm_nt<true, false, false><<<dim3(LQ / 64, Hd / 256, Bq), 256, 0, stream>>>(
      Aq, (long long)LQ * Sd, Sd,
      nullptr, 0, 0, 0,
      DhT, (long long)Hd * Sd, Sd,
      nullptr, QOh + Hd, (long long)LQ * 2 * Hd, 2 * Hd,
      nullptr, Sd);

  // QO^T (per-batch LQ x 2H -> 2H x LQ)
  transpose16<<<dim3(2 * Hd / 32, LQ / 32, Bq), 256, 0, stream>>>(
      QOh, QOT, LQ, 2 * Hd, (long long)LQ * 2 * Hd, (long long)LQ * 2 * Hd);

  // C_D(S,2H) = Ad(S,LQ) x QOT(2H,LQ)^T -> bf16
  gemm_nt<true, false, false><<<dim3(Sd / 64, 2 * Hd / 256, Bq), 256, 0, stream>>>(
      Ad, (long long)Sd * LQ, LQ,
      nullptr, 0, 0, 0,
      QOT, (long long)2 * Hd * LQ, LQ,
      nullptr, Cd, (long long)Sd * 2 * Hd, 2 * Hd,
      nullptr, LQ);

  // doc_output(S,H) = [Cd | Dh](S,3H) x W1(H,3H)^T + b1 -> fp32 d_out
  gemm_nt<false, true, true><<<dim3(Sd / 64, Hd / 256, Bq), 256, 0, stream>>>(
      Cd, (long long)Sd * 2 * Hd, 2 * Hd,
      Dh, (long long)Sd * Hd, Hd, 2 * Hd,
      W1h, 0, 3 * Hd,
      out_doc, nullptr, (long long)Sd * Hd, Hd,
      b1, 3 * Hd);

  // query_out(LQ,H) = QOh(LQ,2H) x W2(H,2H)^T + b2 -> fp32 d_out
  gemm_nt<false, true, false><<<dim3(LQ / 64, Hd / 256, Bq), 256, 0, stream>>>(
      QOh, (long long)LQ * 2 * Hd, 2 * Hd,
      nullptr, 0, 0, 0,
      W2h, 0, 2 * Hd,
      out_q, nullptr, (long long)LQ * Hd, Hd,
      b2, 2 * Hd);

  extract_last<<<(Bq * Hd) / 256, 256, 0, stream>>>(out_doc, doc_lens, out_last);
}
